// TrimMultiheadAttention_12008728560130
// MI455X (gfx1250) — matrix operand order repaired and checked by reading, not compiled, not run on hardware
//
#include <hip/hip_runtime.h>
#include <hip/hip_bf16.h>

// ---------------------------------------------------------------------------
// TrimMultiheadAttention on gfx1250 (MI455X): all GEMMs via v_wmma_f32_16x16x32_bf16
//   E=1024, H=16, D=64, T=2048, B=2
//   in-proj  : 3x [4096,1024]x[1024,1024]^T -> Q/K/V bf16 in [B,H,T,D]
//              (register-blocked: 64x32 per wave, 128x128 per block)
//   attention: chunked causal linear attention, 64-wide chunks, state in WMMA accs
//   out-proj : [4096,1024]x[1024,1024]^T -> fp32 [T,B,E]
// ---------------------------------------------------------------------------

#define E_DIM 1024
#define HEADS 16
#define HDIM 64
#define SEQ_T 2048
#define BATCH_B 2
#define MROWS (SEQ_T * BATCH_B) // 4096

typedef __attribute__((ext_vector_type(16))) __bf16 v16bf;
typedef __attribute__((ext_vector_type(8))) float v8f;

#define DEV static __device__ __forceinline__

DEV v8f wmma_bf16(v16bf a, v16bf b, v8f c) {
  // 8 args: (neg_a, A, neg_b, B, c_mod, C, reuse_a, reuse_b)
  return __builtin_amdgcn_wmma_f32_16x16x32_bf16(false, a, false, b, (short)0, c,
                                                 false, false);
}

// ---- A-matrix (16x32 bf16) loaders: lane L -> row (L&15); K-halves by L>>4 ----
// lane holds K in [k0+8*hi, +7] and [k0+16+8*hi, +7] (ISA 16-bit A layout)
DEV v16bf load_a_bf16(const __bf16* base, int ld, int row0, int k0) {
  const int L = threadIdx.x & 31, m = L & 15, hi = L >> 4;
  const __bf16* p = base + (size_t)(row0 + m) * ld + k0 + hi * 8;
  v16bf r;
#pragma unroll
  for (int i = 0; i < 8; ++i) { r[i] = p[i]; r[i + 8] = p[i + 16]; }
  return r;
}

DEV v16bf load_a_f32(const float* base, int ld, int row0, int k0) {
  const int L = threadIdx.x & 31, m = L & 15, hi = L >> 4;
  const float* p = base + (size_t)(row0 + m) * ld + k0 + hi * 8;
  v16bf r;
#pragma unroll
  for (int i = 0; i < 8; ++i) {
    r[i] = (__bf16)p[i];
    r[i + 8] = (__bf16)p[i + 16];
  }
  return r;
}

// ---- B-matrix (32x16 bf16) loaders: lane L -> col (L&15); K range by L>>4 ----
// B[k][n] = base[(col0+n)*ld + k]  (i.e. B columns are rows of `base` -> x@W^T)
DEV v16bf load_b_bf16(const __bf16* base, int ld, int col0, int k0) {
  const int L = threadIdx.x & 31, n = L & 15, hi = L >> 4;
  const __bf16* p = base + (size_t)(col0 + n) * ld + k0 + hi * 8;
  v16bf r;
#pragma unroll
  for (int i = 0; i < 8; ++i) { r[i] = p[i]; r[i + 8] = p[i + 16]; }
  return r;
}

DEV v16bf load_b_f32(const float* base, int ld, int col0, int k0) {
  const int L = threadIdx.x & 31, n = L & 15, hi = L >> 4;
  const float* p = base + (size_t)(col0 + n) * ld + k0 + hi * 8;
  v16bf r;
#pragma unroll
  for (int i = 0; i < 8; ++i) {
    r[i] = (__bf16)p[i];
    r[i + 8] = (__bf16)p[i + 16];
  }
  return r;
}

// A operand gathered from head-split [B,H,T,D] bf16 (merged-head row-major view)
DEV v16bf load_a_heads(const __bf16* aws, int row0, int k0) {
  const int L = threadIdx.x & 31, hi = L >> 4;
  const int m = row0 + (L & 15); // row of [T*B, E] = t*B + b
  const int t = m >> 1, b = m & 1;
  const int e0 = k0 + hi * 8, e1 = e0 + 16; // both stay inside one 64-wide head
  const __bf16* p0 =
      aws + ((size_t)(b * HEADS + (e0 >> 6)) * SEQ_T + t) * HDIM + (e0 & 63);
  const __bf16* p1 =
      aws + ((size_t)(b * HEADS + (e1 >> 6)) * SEQ_T + t) * HDIM + (e1 & 63);
  v16bf a;
#pragma unroll
  for (int i = 0; i < 8; ++i) { a[i] = p0[i]; a[i + 8] = p1[i]; }
  return a;
}

// ---------------------------------------------------------------------------
// Kernel 1: fused in-projection.  grid(8, 32, 3), block(256) = 8 waves.
// Block tiles 128(M)x128(N); wave computes 64x32 (4x2 WMMA tiles) of
// X[T*B,E] @ W^T + bias, stores bf16 into head-split [B,H,T,D].
// ---------------------------------------------------------------------------
__global__ __launch_bounds__(256) void in_proj_kernel(
    const float* __restrict__ xq, const float* __restrict__ xk,
    const float* __restrict__ xv, const float* __restrict__ wq,
    const float* __restrict__ wk, const float* __restrict__ wv,
    const float* __restrict__ in_bias, __bf16* __restrict__ qws,
    __bf16* __restrict__ kws, __bf16* __restrict__ vws) {
  const int z = blockIdx.z;
  const float* X = (z == 0) ? xq : (z == 1) ? xk : xv;
  const float* W = (z == 0) ? wq : (z == 1) ? wk : wv;
  const float* bptr = in_bias + z * E_DIM;
  __bf16* dst = (z == 0) ? qws : (z == 1) ? kws : vws;

  const int wave = threadIdx.x >> 5;
  const int m0 = blockIdx.y * 128 + (wave >> 2) * 64; // 4 M-tiles
  const int n0 = blockIdx.x * 128 + (wave & 3) * 32;  // 2 N-tiles
  const int L = threadIdx.x & 31, n = L & 15, hi = L >> 4;

  v8f acc[4][2] = {};
#pragma unroll 2
  for (int k0 = 0; k0 < E_DIM; k0 += 32) {
    if (k0 + 64 < E_DIM) { // stream next K panel toward the WGP
      __builtin_prefetch(X + (size_t)(m0 + (L & 15)) * E_DIM + k0 + 64, 0, 3);
      __builtin_prefetch(W + (size_t)(n0 + n) * E_DIM + k0 + 64, 0, 3);
    }
    v16bf a[4], b[2];
#pragma unroll
    for (int i = 0; i < 4; ++i) a[i] = load_a_f32(X, E_DIM, m0 + 16 * i, k0);
#pragma unroll
    for (int j = 0; j < 2; ++j) b[j] = load_b_f32(W, E_DIM, n0 + 16 * j, k0);
#pragma unroll
    for (int i = 0; i < 4; ++i)
#pragma unroll
      for (int j = 0; j < 2; ++j) acc[i][j] = wmma_bf16(a[i], b[j], acc[i][j]);
  }

#pragma unroll
  for (int i = 0; i < 4; ++i)
#pragma unroll
    for (int j = 0; j < 2; ++j) {
      const int e = n0 + 16 * j + n, h = e >> 6, d = e & 63;
      const float bv = bptr[e];
#pragma unroll
      for (int r = 0; r < 8; ++r) {
        const int m = m0 + 16 * i + r + 8 * hi; // row of X = t*B + b ([T,B,E])
        const int t = m >> 1, b2 = m & 1;       // BATCH_B == 2
        dst[((size_t)(b2 * HEADS + h) * SEQ_T + t) * HDIM + d] =
            (__bf16)(acc[i][j][r] + bv);
      }
    }
}

// ---------------------------------------------------------------------------
// Kernel 2: chunked causal linear attention.  grid(32) = one block per (b,h),
// block(256) = 8 waves.  Chunk C=64.  Per chunk:
//   O = scale * ( Q@S_prev + tril(Q@K^T)@V ),   S += K^T@V
// S (64x64 fp32) lives in WMMA accumulators (2 tiles/wave); a bf16 TRANSPOSED
// mirror S_T[dv][dk] in LDS feeds the B operand of Q@S_prev with contiguous
// 16B reads.  K^T,V^T chunks staged transposed in LDS.
// ---------------------------------------------------------------------------
#define PLD 72
__global__ __launch_bounds__(256) void linear_attn_kernel(
    const __bf16* __restrict__ qws, const __bf16* __restrict__ kws,
    const __bf16* __restrict__ vws, __bf16* __restrict__ aws) {
  __shared__ __bf16 S_T[64 * PLD];    // S_T[dv][dk] = S[dk][dv]  (bf16 mirror)
  __shared__ __bf16 P_lds[64 * PLD];  // masked intra-chunk scores (bf16)
  __shared__ __bf16 Kt[64 * PLD];     // Kt[d][c] = K[c][d]
  __shared__ __bf16 Vt[64 * PLD];     // Vt[dv][c] = V[c][dv]

  const int bh = blockIdx.x;
  const __bf16* Qb = qws + (size_t)bh * SEQ_T * HDIM;
  const __bf16* Kb = kws + (size_t)bh * SEQ_T * HDIM;
  const __bf16* Vb = vws + (size_t)bh * SEQ_T * HDIM;
  __bf16* Ob = aws + (size_t)bh * SEQ_T * HDIM;

  const int tid = threadIdx.x;
  const int wave = tid >> 5;
  const int L = tid & 31, n = L & 15, hi = L >> 4;
  const float scale = 0.125f; // 1/sqrt(64)

  for (int i = tid; i < 64 * PLD; i += 256) S_T[i] = (__bf16)0.0f;
  v8f s_acc[2] = {{}, {}};
  __syncthreads();

  for (int c0 = 0; c0 < SEQ_T; c0 += 64) {
    // ---- phase 0: stage transposed K,V chunk into LDS ----
    for (int idx = tid; idx < 64 * 64; idx += 256) {
      const int c = idx >> 6, d = idx & 63;
      Kt[d * PLD + c] = Kb[(size_t)(c0 + c) * HDIM + d];
      Vt[d * PLD + c] = Vb[(size_t)(c0 + c) * HDIM + d];
    }
    __syncthreads();

    v8f o_acc[2];
    // ---- phase 1: P = tril(Q@K^T) -> LDS ;  O = Q@S_prev ----
#pragma unroll
    for (int tt = 0; tt < 2; ++tt) {
      const int t = wave * 2 + tt, ti = t >> 2, tj = t & 3;
      v8f p = {}, o = {};
#pragma unroll
      for (int kk = 0; kk < HDIM; kk += 32) {
        v16bf aq = load_a_bf16(Qb + (size_t)c0 * HDIM, HDIM, ti * 16, kk);
        v16bf bk = load_b_bf16(Kb + (size_t)c0 * HDIM, HDIM, tj * 16, kk);
        p = wmma_bf16(aq, bk, p);
        v16bf bs = load_b_bf16(S_T, PLD, tj * 16, kk); // B[dk][dv] = S_T[dv][dk]
        o = wmma_bf16(aq, bs, o);
      }
#pragma unroll
      for (int r = 0; r < 8; ++r) {
        const int row = ti * 16 + r + 8 * hi, col = tj * 16 + n;
        P_lds[row * PLD + col] = (__bf16)((col <= row) ? p[r] : 0.0f);
      }
      o_acc[tt] = o;
    }
    __syncthreads();

    // ---- phase 2: O += P@V ;  S += K^T@V ; write O, refresh S_T mirror ----
#pragma unroll
    for (int tt = 0; tt < 2; ++tt) {
      const int t = wave * 2 + tt, ti = t >> 2, tj = t & 3;
      v8f o = o_acc[tt];
      v8f s = s_acc[tt];
#pragma unroll
      for (int kk = 0; kk < 64; kk += 32) {
        v16bf bv = load_b_bf16(Vt, PLD, tj * 16, kk); // B[c][dv] (shared)
        v16bf ap = load_a_bf16(P_lds, PLD, ti * 16, kk);
        o = wmma_bf16(ap, bv, o);
        v16bf ak = load_a_bf16(Kt, PLD, ti * 16, kk); // A[dk][c]
        s = wmma_bf16(ak, bv, s);
      }
#pragma unroll
      for (int r = 0; r < 8; ++r) {
        const int m = r + 8 * hi;
        Ob[(size_t)(c0 + ti * 16 + m) * HDIM + tj * 16 + n] =
            (__bf16)(o[r] * scale);
        // transposed bf16 mirror for next chunk's Q@S_prev B operand
        S_T[(tj * 16 + n) * PLD + ti * 16 + m] = (__bf16)s[r];
      }
      s_acc[tt] = s;
    }
    __syncthreads();
  }
}

// ---------------------------------------------------------------------------
// Kernel 3: out-projection.  grid(8, 32), block(256), 64x32 per wave.
// A = merged heads (bf16, gathered from [B,H,T,D]), B = Wo (fp32->bf16),
// out = A@Wo^T + bo, fp32 into [T,B,E].
// ---------------------------------------------------------------------------
__global__ __launch_bounds__(256) void out_proj_kernel(
    const __bf16* __restrict__ aws, const float* __restrict__ Wo,
    const float* __restrict__ bo, float* __restrict__ out) {
  const int wave = threadIdx.x >> 5;
  const int m0 = blockIdx.y * 128 + (wave >> 2) * 64;
  const int n0 = blockIdx.x * 128 + (wave & 3) * 32;
  const int L = threadIdx.x & 31, n = L & 15, hi = L >> 4;

  v8f acc[4][2] = {};
#pragma unroll 2
  for (int k0 = 0; k0 < E_DIM; k0 += 32) {
    if (k0 + 64 < E_DIM)
      __builtin_prefetch(Wo + (size_t)(n0 + n) * E_DIM + k0 + 64, 0, 3);
    v16bf a[4], b[2];
#pragma unroll
    for (int i = 0; i < 4; ++i) a[i] = load_a_heads(aws, m0 + 16 * i, k0);
#pragma unroll
    for (int j = 0; j < 2; ++j) b[j] = load_b_f32(Wo, E_DIM, n0 + 16 * j, k0);
#pragma unroll
    for (int i = 0; i < 4; ++i)
#pragma unroll
      for (int j = 0; j < 2; ++j) acc[i][j] = wmma_bf16(a[i], b[j], acc[i][j]);
  }

#pragma unroll
  for (int i = 0; i < 4; ++i)
#pragma unroll
    for (int j = 0; j < 2; ++j) {
      const float bv = bo[n0 + 16 * j + n];
#pragma unroll
      for (int r = 0; r < 8; ++r) {
        const int m = m0 + 16 * i + r + 8 * hi; // row of [T*B, E] output
        out[(size_t)m * E_DIM + n0 + 16 * j + n] = acc[i][j][r] + bv;
      }
    }
}

// ---------------------------------------------------------------------------
extern "C" void kernel_launch(void* const* d_in, const int* in_sizes, int n_in,
                              void* d_out, int out_size, void* d_ws,
                              size_t ws_size, hipStream_t stream) {
  const float* query = (const float*)d_in[0];
  const float* key_i = (const float*)d_in[1];
  const float* value = (const float*)d_in[2];
  const float* Wq = (const float*)d_in[3];
  const float* Wk = (const float*)d_in[4];
  const float* Wv = (const float*)d_in[5];
  const float* in_bias = (const float*)d_in[6];
  const float* Wo = (const float*)d_in[7];
  const float* bo = (const float*)d_in[8];
  float* out = (float*)d_out;

  const size_t per = (size_t)BATCH_B * HEADS * SEQ_T * HDIM; // 4M elements
  __bf16* qws = (__bf16*)d_ws;
  __bf16* kws = qws + per;
  __bf16* vws = kws + per;
  __bf16* aws = vws + per; // 32 MB total bf16 workspace

  dim3 blk(256);
  in_proj_kernel<<<dim3(E_DIM / 128, MROWS / 128, 3), blk, 0, stream>>>(
      query, key_i, value, Wq, Wk, Wv, in_bias, qws, kws, vws);
  linear_attn_kernel<<<dim3(BATCH_B * HEADS), blk, 0, stream>>>(qws, kws, vws,
                                                                aws);
  out_proj_kernel<<<dim3(E_DIM / 128, MROWS / 128), blk, 0, stream>>>(
      aws, Wo, bo, out);
}
